// CausalAttention_58548994179713
// MI455X (gfx1250) — matrix operand order repaired and checked by reading, not compiled, not run on hardware
//
#include <hip/hip_runtime.h>
#include <hip/hip_bf16.h>
#include <math.h>

typedef __attribute__((ext_vector_type(16))) _Float16 v16h;
typedef __attribute__((ext_vector_type(8)))  _Float16 v8h;
typedef __attribute__((ext_vector_type(8)))  float    v8f;

#define B_DIM   128
#define FEAT    1024
#define NEDGE   32768
#define HID     128
#define K2      256
#define W1_PITCH 264   // padded f16 row stride for LDS bank spread

// ---------- Kernel 1: f32 [128 x 1024] -> f16 transposed [1024 x 128] ----------
__global__ void k_xpose_to_f16T(const float* __restrict__ inImg,
                                const float* __restrict__ inTxt,
                                _Float16* __restrict__ outImgT,
                                _Float16* __restrict__ outTxtT) {
  __shared__ float tile[32][33];
  const float* in  = blockIdx.z ? inTxt  : inImg;
  _Float16*    out = blockIdx.z ? outTxtT : outImgT;
  const int dt = blockIdx.x * 32, bt = blockIdx.y * 32;
  const int tx = threadIdx.x, ty = threadIdx.y;
  #pragma unroll
  for (int q = 0; q < 4; ++q) {
    int b = bt + ty + 8 * q;
    tile[ty + 8 * q][tx] = in[(size_t)b * FEAT + dt + tx];   // [b_local][d_local]
  }
  __syncthreads();
  #pragma unroll
  for (int q = 0; q < 4; ++q) {
    int d = dt + ty + 8 * q;
    out[(size_t)d * B_DIM + bt + tx] = (_Float16)tile[tx][ty + 8 * q];
  }
}

// ---------- Kernel 2: edge MLP via WMMA -> a[E] ----------
// One wave computes 16 edges: D[16x16] per n-tile, K=256 in 8 steps of 32.
__global__ void k_edge_mlp_wmma(const _Float16* __restrict__ imgT,
                                const _Float16* __restrict__ txtT,
                                const int* __restrict__ src,
                                const int* __restrict__ tgt,
                                const float* __restrict__ W1,
                                const float* __restrict__ bias1,
                                const float* __restrict__ wvec2,
                                const float* __restrict__ bias2,
                                float* __restrict__ aOut) {
  extern __shared__ _Float16 w1lds[];   // [HID][W1_PITCH] f16, 67584 B

  const int tid = threadIdx.x;
  // Stage W1 (f32 -> f16) into LDS once per block; reused by all 8 waves.
  for (int idx = tid; idx < HID * K2; idx += 256) {
    int n = idx >> 8;        // row (hidden unit)
    int k = idx & 255;       // col (input feature)
    w1lds[n * W1_PITCH + k] = (_Float16)W1[idx];
  }
  __syncthreads();

  const int wave = tid >> 5;
  const int lane = tid & 31;
  const int m    = lane & 15;     // edge row within tile / n-column
  const int hi   = lane >> 4;     // which half-wave
  const int koff = hi * 8;        // A-fragment K offset for upper half
  const int eb   = (blockIdx.x * 8 + wave) * 16;
  const int e    = eb + m;

  const int sI = src[e];
  const int tI = tgt[e];
  const _Float16* rowImg = imgT + (size_t)sI * B_DIM;
  const _Float16* rowTxt = txtT + (size_t)tI * B_DIM;

  // Preload all 8 A-fragments (pairs row in WMMA A layout).
  v16h afr[8];
  #pragma unroll
  for (int kb = 0; kb < 8; ++kb) {
    const _Float16* base = (kb < 4) ? (rowImg + kb * 32) : (rowTxt + (kb - 4) * 32);
    v8h c0 = *(const v8h*)(base + koff);        // K locals {koff..koff+7}
    v8h c1 = *(const v8h*)(base + koff + 16);   // K locals {koff+16..koff+23}
    afr[kb] = __builtin_shufflevector(c0, c1, 0,1,2,3,4,5,6,7,8,9,10,11,12,13,14,15);
  }

  float spart[8];
  #pragma unroll
  for (int r = 0; r < 8; ++r) spart[r] = 0.f;

  #pragma unroll
  for (int nt = 0; nt < 8; ++nt) {
    const int n = nt * 16 + m;                  // this lane's B column / hidden idx
    const _Float16* wrow = w1lds + n * W1_PITCH;
    v8f acc = {};
    #pragma unroll
    for (int kb = 0; kb < 8; ++kb) {
      const int ks = kb * 32 + hi * 8;          // lanes 0-15: K 0..7,16..23; 16-31: K 8..15,24..31
      v8h c0 = *(const v8h*)(wrow + ks);
      v8h c1 = *(const v8h*)(wrow + ks + 16);
      v16h bfr = __builtin_shufflevector(c0, c1, 0,1,2,3,4,5,6,7,8,9,10,11,12,13,14,15);
      acc = __builtin_amdgcn_wmma_f32_16x16x32_f16(
          /*neg_a=*/false, afr[kb], /*neg_b=*/false, bfr,
          /*c_mod=*/(short)0, acc, /*reuse_a=*/false, /*reuse_b=*/false);
    }
    const float bv = bias1[n];
    const float wv = wvec2[n];
    #pragma unroll
    for (int r = 0; r < 8; ++r) {               // rows m = 8*hi + r
      float h = acc[r] + bv;
      h = h > 0.f ? h : 0.f;
      spart[r] += h * wv;                        // partial over this lane's n's
    }
  }

  // Reduce over the 16 n-columns held by each half-wave.
  #pragma unroll
  for (int dlt = 1; dlt < 16; dlt <<= 1) {
    #pragma unroll
    for (int r = 0; r < 8; ++r)
      spart[r] += __shfl_xor(spart[r], dlt, 32);
  }

  if (m == 0) {
    const float bb = bias2[0];
    #pragma unroll
    for (int r = 0; r < 8; ++r) {
      float x = spart[r] + bb;
      aOut[eb + hi * 8 + r] = 1.0f / (1.0f + __expf(-x));
    }
  }
}

// ---------- Kernel 3: scatter-accumulate into transposed accumulators ----------
__global__ void k_edge_scatter(const float* __restrict__ aE,
                               const int* __restrict__ src,
                               const int* __restrict__ tgt,
                               const _Float16* __restrict__ imgT,
                               const _Float16* __restrict__ txtT,
                               float* __restrict__ accImgT,
                               float* __restrict__ accTxtT) {
  const int t = blockIdx.x * blockDim.x + threadIdx.x;
  const int e = t >> 7;
  const int b = t & 127;
  const float a = aE[e];
  const int sI = src[e];
  const int tI = tgt[e];
  const float tv = (float)txtT[(size_t)tI * B_DIM + b];
  const float iv = (float)imgT[(size_t)sI * B_DIM + b];
  atomicAdd(accImgT + (size_t)sI * B_DIM + b, a * tv);  // contiguous across lanes
  atomicAdd(accTxtT + (size_t)tI * B_DIM + b, a * iv);
}

// ---------- Kernel 4: transpose f32 [1024 x 128] -> output [128 x 1024] ----------
__global__ void k_xposeT_to_out(const float* __restrict__ accImgT,
                                const float* __restrict__ accTxtT,
                                float* __restrict__ out) {
  __shared__ float tile[32][33];
  const float* in = blockIdx.z ? accTxtT : accImgT;
  float* o = out + (size_t)blockIdx.z * B_DIM * FEAT;
  const int dt = blockIdx.x * 32, bt = blockIdx.y * 32;
  const int tx = threadIdx.x, ty = threadIdx.y;
  #pragma unroll
  for (int q = 0; q < 4; ++q) {
    int d = dt + ty + 8 * q;
    tile[ty + 8 * q][tx] = in[(size_t)d * B_DIM + bt + tx];  // [d_local][b_local]
  }
  __syncthreads();
  #pragma unroll
  for (int q = 0; q < 4; ++q) {
    int b = bt + ty + 8 * q;
    o[(size_t)b * FEAT + dt + tx] = tile[tx][ty + 8 * q];
  }
}

extern "C" void kernel_launch(void* const* d_in, const int* in_sizes, int n_in,
                              void* d_out, int out_size, void* d_ws, size_t ws_size,
                              hipStream_t stream) {
  const float* img   = (const float*)d_in[0];
  const float* txt   = (const float*)d_in[1];
  const int*   src   = (const int*)d_in[2];
  const int*   tgt   = (const int*)d_in[3];
  const float* W1    = (const float*)d_in[4];
  const float* bias1 = (const float*)d_in[5];
  const float* wvec2 = (const float*)d_in[6];
  const float* bias2 = (const float*)d_in[7];
  float* out = (float*)d_out;

  char* ws = (char*)d_ws;
  _Float16* imgT    = (_Float16*)(ws);                 // 256 KB
  _Float16* txtT    = (_Float16*)(ws + 262144);        // 256 KB
  float*    aE      = (float*)(ws + 524288);           // 128 KB
  float*    accImgT = (float*)(ws + 655360);           // 512 KB
  float*    accTxtT = (float*)(ws + 1179648);          // 512 KB

  hipMemsetAsync(accImgT, 0, 2u * 524288u, stream);    // zero both accumulators

  dim3 tb(32, 8);
  k_xpose_to_f16T<<<dim3(FEAT / 32, B_DIM / 32, 2), tb, 0, stream>>>(
      img, txt, imgT, txtT);

  const size_t ldsBytes = (size_t)HID * W1_PITCH * sizeof(_Float16);  // 67584 B
  k_edge_mlp_wmma<<<NEDGE / 128, 256, ldsBytes, stream>>>(
      imgT, txtT, src, tgt, W1, bias1, wvec2, bias2, aE);

  k_edge_scatter<<<(NEDGE * B_DIM) / 256, 256, 0, stream>>>(
      aE, src, tgt, imgT, txtT, accImgT, accTxtT);

  k_xposeT_to_out<<<dim3(FEAT / 32, B_DIM / 32, 2), tb, 0, stream>>>(
      accImgT, accTxtT, out);
}
